// CausalSelfAttention_78984448573479
// MI455X (gfx1250) — matrix operand order repaired and checked by reading, not compiled, not run on hardware
//
#include <hip/hip_runtime.h>
#include <stdint.h>

// ---------------- problem constants ----------------
constexpr int Bn  = 2;
constexpr int Tn  = 2048;
constexpr int Hn  = 2048;
constexpr int NHn = 16;
constexpr int HSn = 128;           // head size
constexpr int Mn  = Bn * Tn;       // 4096 flattened rows

// ---------------- types ----------------
typedef __bf16 bf16_t;
typedef __attribute__((ext_vector_type(16))) __bf16 v16bf;
typedef __attribute__((ext_vector_type(8)))  float  v8f;

union FragU { v16bf v; uint32_t u[8]; };

// ---------------- TDM availability ----------------
#if defined(__has_builtin)
#if __has_builtin(__builtin_amdgcn_tensor_load_to_lds) && \
    __has_builtin(__builtin_amdgcn_s_wait_tensorcnt)
#define USE_TDM 1
#endif
#endif
#ifndef USE_TDM
#define USE_TDM 0
#endif

#if USE_TDM
typedef unsigned int v4u  __attribute__((ext_vector_type(4)));
typedef int          v8i32 __attribute__((ext_vector_type(8)));
typedef int          v4i32 __attribute__((ext_vector_type(4)));

// Issue a 2-D TENSOR_LOAD_TO_LDS per CDNA5 ISA §8.3/8.4.
//   tensor: td1 rows x td0 cols (dim0 fastest, "data_size units" = 2B elems)
//   tile  : tile1 rows x tile0 cols, row stride = stride0 elems
__device__ __forceinline__ void tdm_load_2d(const void* gsrc, uint32_t lds_addr,
                                            uint32_t td0, uint32_t td1,
                                            uint32_t tile0, uint32_t tile1,
                                            uint32_t stride0) {
  uint64_t ga = (uint64_t)(uintptr_t)gsrc;
  v4u g0;
  g0[0] = 1u;                                       // count=1 (valid user D#)
  g0[1] = lds_addr;                                 // lds_addr (bytes)
  g0[2] = (uint32_t)ga;                             // global_addr[31:0]
  g0[3] = (uint32_t)((ga >> 32) & 0x1FFFFFFu)       // global_addr[56:32]
          | (2u << 30);                             // type = 2 ("image")
  v8i32 g1;
  g1[0] = (int)(1u << 16);                          // wg_mask=0, data_size=1 (2B)
  g1[1] = (int)((td0 & 0xFFFFu) << 16);             // tensor_dim0 @ bit 48
  g1[2] = (int)(((td0 >> 16) & 0xFFFFu) | ((td1 & 0xFFFFu) << 16));
  g1[3] = (int)(((td1 >> 16) & 0xFFFFu) | ((tile0 & 0xFFFFu) << 16));
  g1[4] = (int)(tile1 & 0xFFFFu);                   // tile_dim1; tile_dim2=0
  g1[5] = (int)stride0;                             // tensor_dim0_stride[31:0]
  g1[6] = 0;                                        // stride hi / dim1_stride
  g1[7] = 0;
  v4i32 z4 = {};
#if __clang_major__ >= 23
  v8i32 z8 = {};
  __builtin_amdgcn_tensor_load_to_lds(g0, g1, z4, z4, z8, 0);
#else
  __builtin_amdgcn_tensor_load_to_lds(g0, g1, z4, z4, 0);
#endif
}
#endif  // USE_TDM

__device__ __forceinline__ bf16_t f2bf(float x) {
  union { float f; uint32_t u; } v; v.f = x;
  uint32_t r = (v.u + 0x7FFFu + ((v.u >> 16) & 1u)) >> 16;
  union { uint16_t u; bf16_t b; } o; o.u = (uint16_t)r;
  return o.b;
}
__device__ __forceinline__ float bf2f(bf16_t x) {
  union { uint16_t u; bf16_t b; } i; i.b = x;
  union { float f; uint32_t u; } o; o.u = ((uint32_t)i.u) << 16;
  return o.f;
}

// WMMA bf16 16x16x32, f32 accumulate (codegen-confirmed builtin)
__device__ __forceinline__ v8f wmma_bf16(v16bf a, v16bf b, v8f c) {
  return __builtin_amdgcn_wmma_f32_16x16x32_bf16(
      false, a, false, b, (short)0, c, false, false);
}

// A-matrix 16x32 bf16 fragment (this code's contiguous k order, same as B): lane = M row (lane&15),
// VGPR j : K = 2j + 16*half.
__device__ __forceinline__ v16bf load_frag_a(const bf16_t* __restrict__ src,
                                             int ld, int m0, int k0, int lane) {
  FragU f;
  const int row  = m0 + (lane & 15);
  const int half = lane >> 4;
  const bf16_t* p = src + (size_t)row * ld + k0;
#pragma unroll
  for (int j = 0; j < 8; ++j) {
    int kk = 2 * j + 16 * half;
    f.u[j] = *reinterpret_cast<const uint32_t*>(p + kk);
  }
  return f.v;
}

// B-matrix 32x16 bf16 fragment: lane = N col (lane&15), VGPR j : K = 2j+16*half.
// Element (k,n) = src[n*ld + k] ("n-major" storage) -> one dword per VGPR.
__device__ __forceinline__ v16bf load_frag_b(const bf16_t* __restrict__ src,
                                             int ld, int k0, int n0, int lane) {
  FragU f;
  const int col  = n0 + (lane & 15);
  const int half = lane >> 4;
  const bf16_t* p = src + (size_t)col * ld + k0 + 16 * half;
#pragma unroll
  for (int j = 0; j < 8; ++j)
    f.u[j] = *reinterpret_cast<const uint32_t*>(p + 2 * j);
  return f.v;
}

// ---------------- kernel 1: fp32 -> bf16 ----------------
__global__ void cvt_bf16_kernel(const float* __restrict__ in,
                                bf16_t* __restrict__ out, int n) {
  int i = blockIdx.x * blockDim.x + threadIdx.x;
  if (i < n) out[i] = f2bf(in[i]);
}

// ---------------- kernel 2: W[k][n] -> Wt[n][k] bf16 ----------------
__global__ void cvt_wT_kernel(const float* __restrict__ w,
                              bf16_t* __restrict__ wT) {
  int idx = blockIdx.x * blockDim.x + threadIdx.x;   // over Hn*Hn
  int k = idx >> 11;        // Hn = 2048
  int n = idx & (Hn - 1);
  wT[(size_t)n * Hn + k] = f2bf(w[idx]);
}

// ---------------- kernel 3: fused QKV projection GEMM ----------------
// grid (Mn/128, Hn/128, 3); block 256 = 8 waves; wave tile 32x64.
// which==0 -> Q [B,NH,T,HS], 1 -> K [B,NH,T,HS], 2 -> V^T [B,NH,HS,T]
__global__ __launch_bounds__(256) void qkv_gemm_kernel(
    const bf16_t* __restrict__ xb,
    const bf16_t* __restrict__ wqT, const bf16_t* __restrict__ wkT,
    const bf16_t* __restrict__ wvT,
    const float* __restrict__ bq, const float* __restrict__ bk,
    const float* __restrict__ bv,
    bf16_t* __restrict__ qout, bf16_t* __restrict__ kout,
    bf16_t* __restrict__ vTout) {
  const int which = blockIdx.z;
  const bf16_t* wT = (which == 0) ? wqT : (which == 1) ? wkT : wvT;
  const float* bias = (which == 0) ? bq : (which == 1) ? bk : bv;

  const int lane = threadIdx.x & 31;
  const int wave = threadIdx.x >> 5;
  const int wm = wave & 3;            // 4 waves along M
  const int wn = wave >> 2;           // 2 waves along N
  const int m_base = blockIdx.x * 128 + wm * 32;
  const int n_base = blockIdx.y * 128 + wn * 64;

  v8f acc[2][4] = {};
#pragma unroll 4
  for (int k0 = 0; k0 < Hn; k0 += 32) {
    v16bf a0 = load_frag_a(xb, Hn, m_base,      k0, lane);
    v16bf a1 = load_frag_a(xb, Hn, m_base + 16, k0, lane);
#pragma unroll
    for (int nt = 0; nt < 4; ++nt) {
      v16bf bfr = load_frag_b(wT, Hn, k0, n_base + nt * 16, lane);
      acc[0][nt] = wmma_bf16(a0, bfr, acc[0][nt]);
      acc[1][nt] = wmma_bf16(a1, bfr, acc[1][nt]);
    }
  }

  const int half = lane >> 4, nloc = lane & 15;
#pragma unroll
  for (int nt = 0; nt < 4; ++nt) {
    const int n = n_base + nt * 16 + nloc;
    const float bval = bias[n];
    const int h = n >> 7;             // / HSn
    const int d = n & (HSn - 1);
#pragma unroll
    for (int mt = 0; mt < 2; ++mt) {
#pragma unroll
      for (int r = 0; r < 8; ++r) {
        float val = acc[mt][nt][r] + bval;
        int m = m_base + mt * 16 + r + 8 * half;
        int b = m >> 11;              // / Tn
        int t = m & (Tn - 1);
        int bh = b * NHn + h;
        if (which == 2)
          vTout[((size_t)bh * HSn + d) * Tn + t] = f2bf(val);
        else if (which == 1)
          kout[((size_t)bh * Tn + t) * HSn + d] = f2bf(val);
        else
          qout[((size_t)bh * Tn + t) * HSn + d] = f2bf(val);
      }
    }
  }
}

// ---------------- kernel 4: RoPE in place on Q and K ----------------
__global__ void rope_kernel(bf16_t* __restrict__ q, bf16_t* __restrict__ k) {
  int idx = blockIdx.x * blockDim.x + threadIdx.x;  // Bn*NHn*Tn*(HSn/2)
  int i  = idx & 63;                 // HSn/2 = 64
  int t  = (idx >> 6) & (Tn - 1);
  int bh = idx >> 17;
  float inv_freq = __powf(10000.0f, -((float)(2 * i)) / (float)HSn);
  float ang = (float)t * inv_freq;
  float s, c;
  __sincosf(ang, &s, &c);
  size_t off = ((size_t)bh * Tn + t) * HSn + 2 * i;
  {
    float x1 = bf2f(q[off]), x2 = bf2f(q[off + 1]);
    q[off]     = f2bf(x1 * c - x2 * s);
    q[off + 1] = f2bf(x1 * s + x2 * c);
  }
  {
    float x1 = bf2f(k[off]), x2 = bf2f(k[off + 1]);
    k[off]     = f2bf(x1 * c - x2 * s);
    k[off + 1] = f2bf(x1 * s + x2 * c);
  }
}

// ---------------- kernel 5: attention ----------------
// grid (Tn/64, Bn*NHn), block 128 (4 waves x 16 query rows).
// TDM path: wave0 DMAs the 64x128 K tile and 128x64 V^T tile of kv-step j+1
// into double-buffered LDS while all waves run WMMAs on tile j; sync with
// s_wait_tensorcnt + one workgroup barrier per iteration.
// Reference semantics preserved: softmax denominator over ALL keys, causal
// zeroing applied to the numerator only.
//
// dynamic LDS layout (bf16 elems):
//   [0,4096)            per-wave P bounce tiles (4 x 16x64)
//   [4096,20480)        K tiles  (2 x 64 tok x 128 dim), double buffered
//   [20480,36864)       V tiles  (2 x 128 dim x 64 tok), double buffered
#define ATTN_SMEM_BYTES ((4 * 16 * 64 + 4 * 64 * 128) * 2)

__global__ __launch_bounds__(128) void attn_kernel(
    const bf16_t* __restrict__ q, const bf16_t* __restrict__ k,
    const bf16_t* __restrict__ vT, bf16_t* __restrict__ attnout) {
  extern __shared__ bf16_t smem[];

  const int lane = threadIdx.x & 31;
  const int wave = threadIdx.x >> 5;
  const int bh = blockIdx.y;
  const int qbase = blockIdx.x * 64 + wave * 16;
  const int half = lane >> 4, nloc = lane & 15;

  bf16_t* lds_p = smem + wave * (16 * 64);

  const bf16_t* qh = q  + (size_t)bh * Tn * HSn;
  const bf16_t* kh = k  + (size_t)bh * Tn * HSn;
  const bf16_t* vh = vT + (size_t)bh * HSn * Tn;

  // Q fragments for this wave's 16 rows, K = 128 -> 4 fragments
  v16bf aq[4];
#pragma unroll
  for (int kt = 0; kt < 4; ++kt)
    aq[kt] = load_frag_a(qh, HSn, qbase, kt * 32, lane);

  v8f acc[8] = {};
  float rowmax[8], rowsum[8];
#pragma unroll
  for (int r = 0; r < 8; ++r) { rowmax[r] = -3.0e38f; rowsum[r] = 0.0f; }

  const float scale = 0.08838834764831845f;  // 1/sqrt(128)
  constexpr int NJ = Tn / 64;

#if USE_TDM
  // NOTE: compute LDS double-buffer pointers with runtime arithmetic only --
  // constant-initialized pointer arrays of LDS addresses become static
  // initializers with addrspacecast exprs that ld.lld rejects.
  constexpr int KV_TILE_ELEMS = 64 * 128;     // 16 KB each
  bf16_t* kbase = smem + 4096;
  bf16_t* vbase = smem + 4096 + 2 * KV_TILE_ELEMS;

  // prologue: stage tile 0
  if (wave == 0) {
    tdm_load_2d(kh, (uint32_t)(uintptr_t)kbase, HSn, Tn, HSn, 64, HSn);
    tdm_load_2d(vh, (uint32_t)(uintptr_t)vbase, Tn, HSn, 64, HSn, Tn);
    __builtin_amdgcn_s_wait_tensorcnt(0);
  }
  __syncthreads();
#endif

  for (int j = 0; j < NJ; ++j) {             // full sweep (see note above)
    const int kv0 = j * 64;

#if USE_TDM
    if (wave == 0 && j + 1 < NJ) {           // prefetch tile j+1 via TDM
      const int kv1 = (j + 1) * 64;
      bf16_t* knext = kbase + ((j + 1) & 1) * KV_TILE_ELEMS;
      bf16_t* vnext = vbase + ((j + 1) & 1) * KV_TILE_ELEMS;
      tdm_load_2d(kh + (size_t)kv1 * HSn, (uint32_t)(uintptr_t)knext,
                  HSn, Tn, HSn, 64, HSn);
      tdm_load_2d(vh + kv1, (uint32_t)(uintptr_t)vnext,
                  Tn, HSn, 64, HSn, Tn);
    }
    const bf16_t* ksrc = kbase + (j & 1) * KV_TILE_ELEMS;
    const bf16_t* vsrc = vbase + (j & 1) * KV_TILE_ELEMS;
    const int vld = 64;
    const int kvloc = 0;                     // tile-local base
#else
    const bf16_t* ksrc = kh;
    const bf16_t* vsrc = vh;
    const int vld = Tn;
    const int kvloc = kv0;
#endif

    // S = Q K^T  (16x64), 16 WMMAs
    v8f s[4] = {};
#pragma unroll
    for (int nt = 0; nt < 4; ++nt) {
#pragma unroll
      for (int kt = 0; kt < 4; ++kt) {
        v16bf bk = load_frag_b(ksrc, HSn, kt * 32, kvloc + nt * 16, lane);
        s[nt] = wmma_bf16(aq[kt], bk, s[nt]);
      }
    }

    // online softmax statistics (row = r + 8*half, cols across 16 lanes)
    float alpha[8];
#pragma unroll
    for (int r = 0; r < 8; ++r) {
      const int qm = qbase + r + 8 * half;
      float lm = -3.0e38f;
#pragma unroll
      for (int nt = 0; nt < 4; ++nt) {
        float sv = s[nt][r] * scale;
        s[nt][r] = sv;
        lm = fmaxf(lm, sv);
      }
#pragma unroll
      for (int msk = 1; msk < 16; msk <<= 1)
        lm = fmaxf(lm, __shfl_xor(lm, msk, 32));
      float mnew = fmaxf(rowmax[r], lm);
      alpha[r] = __expf(rowmax[r] - mnew);
      rowmax[r] = mnew;
      float psum = 0.0f;
#pragma unroll
      for (int nt = 0; nt < 4; ++nt) {
        float pv = __expf(s[nt][r] - mnew);
        int col = kv0 + nt * 16 + nloc;      // global key index
        s[nt][r] = (col <= qm) ? pv : 0.0f;  // mask numerator only
        psum += pv;                          // denominator over ALL keys
      }
#pragma unroll
      for (int msk = 1; msk < 16; msk <<= 1)
        psum += __shfl_xor(psum, msk, 32);
      rowsum[r] = rowsum[r] * alpha[r] + psum;
    }

    // rescale accumulator
#pragma unroll
    for (int f = 0; f < 8; ++f)
#pragma unroll
      for (int r = 0; r < 8; ++r) acc[f][r] *= alpha[r];

    // bounce P through LDS: C-layout -> A-layout (wave-local, DS in-order)
#pragma unroll
    for (int nt = 0; nt < 4; ++nt)
#pragma unroll
      for (int r = 0; r < 8; ++r)
        lds_p[(r + 8 * half) * 64 + nt * 16 + nloc] = f2bf(s[nt][r]);
    asm volatile("s_wait_dscnt 0" ::: "memory");
    v16bf ap0 = load_frag_a(lds_p, 64, 0, 0,  lane);
    v16bf ap1 = load_frag_a(lds_p, 64, 0, 32, lane);
    asm volatile("" ::: "memory");

    // O += P V  (16x128), 16 WMMAs
#pragma unroll
    for (int ot = 0; ot < 8; ++ot) {
      v16bf bv0 = load_frag_b(vsrc, vld, kvloc,      ot * 16, lane);
      v16bf bv1 = load_frag_b(vsrc, vld, kvloc + 32, ot * 16, lane);
      acc[ot] = wmma_bf16(ap0, bv0, acc[ot]);
      acc[ot] = wmma_bf16(ap1, bv1, acc[ot]);
    }

#if USE_TDM
    if (wave == 0 && j + 1 < NJ)
      __builtin_amdgcn_s_wait_tensorcnt(0);  // tile j+1 landed in LDS
    __syncthreads();                         // publish to all waves
#endif
  }

  // epilogue: O / rowsum -> attn buffer [B, T, H] as bf16
  const int b = bh >> 4;              // / NHn
  const int h = bh & (NHn - 1);
#pragma unroll
  for (int ot = 0; ot < 8; ++ot) {
    const int d = ot * 16 + nloc;
#pragma unroll
    for (int r = 0; r < 8; ++r) {
      float val = acc[ot][r] / rowsum[r];
      int t = qbase + r + 8 * half;
      attnout[((size_t)(b * Tn + t)) * Hn + h * HSn + d] = f2bf(val);
    }
  }
}

// ---------------- kernel 6: output projection, fp32 result ----------------
__global__ __launch_bounds__(256) void proj_gemm_kernel(
    const bf16_t* __restrict__ ab, const bf16_t* __restrict__ wT,
    const float* __restrict__ bias, float* __restrict__ out) {
  const int lane = threadIdx.x & 31;
  const int wave = threadIdx.x >> 5;
  const int wm = wave & 3;
  const int wn = wave >> 2;
  const int m_base = blockIdx.x * 128 + wm * 32;
  const int n_base = blockIdx.y * 128 + wn * 64;

  v8f acc[2][4] = {};
#pragma unroll 4
  for (int k0 = 0; k0 < Hn; k0 += 32) {
    v16bf a0 = load_frag_a(ab, Hn, m_base,      k0, lane);
    v16bf a1 = load_frag_a(ab, Hn, m_base + 16, k0, lane);
#pragma unroll
    for (int nt = 0; nt < 4; ++nt) {
      v16bf bfr = load_frag_b(wT, Hn, k0, n_base + nt * 16, lane);
      acc[0][nt] = wmma_bf16(a0, bfr, acc[0][nt]);
      acc[1][nt] = wmma_bf16(a1, bfr, acc[1][nt]);
    }
  }

  const int half = lane >> 4, nloc = lane & 15;
#pragma unroll
  for (int nt = 0; nt < 4; ++nt) {
    const int n = n_base + nt * 16 + nloc;
    const float bval = bias[n];
#pragma unroll
    for (int mt = 0; mt < 2; ++mt) {
#pragma unroll
      for (int r = 0; r < 8; ++r) {
        int m = m_base + mt * 16 + r + 8 * half;
        out[(size_t)m * Hn + n] = acc[mt][nt][r] + bval;
      }
    }
  }
}

// ---------------- host: launch ----------------
extern "C" void kernel_launch(void* const* d_in, const int* in_sizes, int n_in,
                              void* d_out, int out_size, void* d_ws,
                              size_t ws_size, hipStream_t stream) {
  (void)in_sizes; (void)n_in; (void)out_size; (void)ws_size;
  const float* x  = (const float*)d_in[0];
  const float* Wq = (const float*)d_in[1];
  const float* bq = (const float*)d_in[2];
  const float* Wk = (const float*)d_in[3];
  const float* bk = (const float*)d_in[4];
  const float* Wv = (const float*)d_in[5];
  const float* bv = (const float*)d_in[6];
  const float* Wo = (const float*)d_in[7];
  const float* bo = (const float*)d_in[8];
  float* out = (float*)d_out;

  char* p = (char*)d_ws;
  auto take = [&](size_t bytes) -> char* {
    char* r = p;
    p += (bytes + 255) & ~(size_t)255;
    return r;
  };
  bf16_t* xb   = (bf16_t*)take((size_t)Mn * Hn * 2);
  bf16_t* wqT  = (bf16_t*)take((size_t)Hn * Hn * 2);
  bf16_t* wkT  = (bf16_t*)take((size_t)Hn * Hn * 2);
  bf16_t* wvT  = (bf16_t*)take((size_t)Hn * Hn * 2);
  bf16_t* woT  = (bf16_t*)take((size_t)Hn * Hn * 2);
  bf16_t* qb   = (bf16_t*)take((size_t)Bn * NHn * Tn * HSn * 2);
  bf16_t* kb   = (bf16_t*)take((size_t)Bn * NHn * Tn * HSn * 2);
  bf16_t* vTb  = (bf16_t*)take((size_t)Bn * NHn * HSn * Tn * 2);
  bf16_t* attn = (bf16_t*)take((size_t)Mn * Hn * 2);

  const int n_x = Mn * Hn;
  cvt_bf16_kernel<<<n_x / 256, 256, 0, stream>>>(x, xb, n_x);

  const int n_w = Hn * Hn;
  cvt_wT_kernel<<<n_w / 256, 256, 0, stream>>>(Wq, wqT);
  cvt_wT_kernel<<<n_w / 256, 256, 0, stream>>>(Wk, wkT);
  cvt_wT_kernel<<<n_w / 256, 256, 0, stream>>>(Wv, wvT);
  cvt_wT_kernel<<<n_w / 256, 256, 0, stream>>>(Wo, woT);

  dim3 gq(Mn / 128, Hn / 128, 3);
  qkv_gemm_kernel<<<gq, 256, 0, stream>>>(xb, wqT, wkT, wvT, bq, bk, bv,
                                          qb, kb, vTb);

  const int n_rope = Bn * NHn * Tn * (HSn / 2);
  rope_kernel<<<n_rope / 256, 256, 0, stream>>>(qb, kb);

  attn_kernel<<<dim3(Tn / 64, Bn * NHn), 128, ATTN_SMEM_BYTES, stream>>>(
      qb, kb, vTb, attn);

  proj_gemm_kernel<<<dim3(Mn / 128, Hn / 128), 256, 0, stream>>>(attn, woT,
                                                                 bo, out);
}
